// actor_4999341933053
// MI455X (gfx1250) — matrix operand order repaired and checked by reading, not compiled, not run on hardware
//
#include <hip/hip_runtime.h>
#include <hip/hip_bf16.h>

typedef __attribute__((ext_vector_type(16))) _Float16 v16h;
typedef __attribute__((ext_vector_type(8)))  _Float16 v8h;
typedef __attribute__((ext_vector_type(4)))  _Float16 v4h;
typedef __attribute__((ext_vector_type(8)))  float    v8f;
typedef __attribute__((ext_vector_type(4)))  unsigned int v4u;
typedef __attribute__((ext_vector_type(8)))  int      v8i;
typedef __attribute__((ext_vector_type(4)))  int      v4i;

#define ZZ     64
#define LDH    72                      // padded row stride in f16 (36 dwords, conflict-free)
#define NWAVES 4
#define WROWS  16
#define BROWS  (NWAVES * WROWS)        // 64 rows per block
#define WELEM  (3 * ZZ * LDH)          // f16 elements per zone weight block (13824 = 27648 B)

#if __has_builtin(__builtin_amdgcn_tensor_load_to_lds) && __has_builtin(__builtin_amdgcn_s_wait_tensorcnt)
#define HAVE_TDM 1
#endif

__device__ __forceinline__ float fast_tanh(float x) {
#if __has_builtin(__builtin_amdgcn_tanhf)
    return __builtin_amdgcn_tanhf(x);          // v_tanh_f32 (gfx1250 trans op)
#else
    float e = __expf(2.0f * x);                // exact at +-inf
    return 1.0f - 2.0f / (e + 1.0f);
#endif
}
__device__ __forceinline__ float fast_sigmoid(float x) {
    return 1.0f / (1.0f + __expf(-x));
}

// XOR-butterfly add via ds_swizzle_b32 (group-of-32 mode: and=0x1F, or=0, xor=PAT>>10).
// Single DS op per exchange; no index VALU, stays inside the 16-lane half for xor<=8.
template <int PAT>
__device__ __forceinline__ float bfly_add(float v) {
    return v + __int_as_float(__builtin_amdgcn_ds_swizzle(__float_as_int(v), PAT));
}
#define SWZ_X1 0x041F
#define SWZ_X2 0x081F
#define SWZ_X4 0x101F
#define SWZ_X8 0x201F

#if defined(HAVE_TDM)
// DMA one pre-padded f16 zone-weight block (27648 B) global -> LDS via the TDM.
// Described as a 2-D tile: data_size=8B, dim0 = 3456 QWORDs, 1 row. Groups 2/3 zero.
// amdgpu-toolchain (clang-23) arity: (v4u, v8i, v4i, v4i, v8i, i32 cpol).
__device__ __forceinline__ void tdm_load_weights(const _Float16* gsrc, _Float16* ldst) {
    unsigned lds_off = (unsigned)(uintptr_t)ldst;          // flat LDS addr[31:0] = LDS offset
    unsigned long long ga = (unsigned long long)(uintptr_t)gsrc;
    v4u g0 = { 1u,                                          // count=1, user mode
               lds_off,                                     // lds_addr
               (unsigned)(ga & 0xFFFFFFFFu),                // global_addr[31:0]
               (unsigned)((ga >> 32) & 0x1FFFFFFu) | (2u << 30) }; // addr[56:32] | type=2
    v8i g1 = { (int)(3u << 16),                             // wg_mask=0, data_size=3 (8B)
               (int)((3456u & 0xFFFFu) << 16),              // tensor_dim0[15:0]
               (int)((3456u >> 16) | (1u << 16)),           // tensor_dim0[31:16] | tensor_dim1[15:0]=1
               (int)(3456u << 16),                          // tensor_dim1[31:16]=0 | tile_dim0=3456
               (int)(1u),                                   // tile_dim1=1, tile_dim2=0
               (int)(3456u),                                // tensor_dim0_stride[31:0]
               0, 0 };                                      // stride hi / dim1 stride = 0
    v4i gz4 = {0, 0, 0, 0};
    v8i gz8 = {0, 0, 0, 0, 0, 0, 0, 0};
    __builtin_amdgcn_tensor_load_to_lds(g0, g1, gz4, gz4, gz8, 0);
}
#else
__device__ __forceinline__ void coop_copy_weights(const _Float16* src, _Float16* dst, int tid) {
    const unsigned long long* s = (const unsigned long long*)src;
    unsigned long long*       d = (unsigned long long*)dst;
    for (int i = tid; i < WELEM / 4; i += NWAVES * 32) d[i] = s[i];   // 8B chunks
}
#endif

// Stage h tile (C/D-layout registers) into wave-private LDS as f16, optional mask scale.
template <bool MASK>
__device__ __forceinline__ void stage_h(_Float16* hst, const v8f (&h)[4],
                                        const float (&mk)[4], int l, int half) {
#pragma unroll
    for (int t = 0; t < 4; ++t) {
#pragma unroll
        for (int r = 0; r < 8; ++r) {
            int m = r + 8 * half;              // C-layout: VGPR r holds M=r / M=r+8
            float v = h[t][r];
            if (MASK) v *= mk[t];
            hst[m * LDH + t * 16 + l] = (_Float16)v;
        }
    }
    // LDS is in-order per wave; fence compiler + drain DS so cross-lane reads are safe.
    asm volatile("s_wait_dscnt 0" ::: "memory");
}

// [16x64] @ W^T for this wave: 8x v_wmma_f32_16x16x32_f16 (4 N-tiles x 2 K-chunks).
__device__ __forceinline__ void wave_gemm(const _Float16* hst, const _Float16* wm,
                                          int l, int half, const float (&bias)[4],
                                          v8f (&acc)[4]) {
#pragma unroll
    for (int t = 0; t < 4; ++t) {
#pragma unroll
        for (int i = 0; i < 8; ++i) acc[t][i] = bias[t];
    }
    const int m = l;                           // A layout: both lane halves map to rows 0..15
#pragma unroll
    for (int c = 0; c < 2; ++c) {              // K chunks of 32
        union { v16h v; v8h p[2]; } a;
        const int abase = c * 32 + half * 8;   // K = abase..+7 and abase+16..+23
        a.p[0] = *(const v8h*)(hst + m * LDH + abase);
        a.p[1] = *(const v8h*)(hst + m * LDH + abase + 16);
#pragma unroll
        for (int t = 0; t < 4; ++t) {          // N tiles of 16
            union { v16h v; v8h p[2]; } b;
            const int N  = t * 16 + l;
            const int k0 = c * 32 + half * 8;  // B: K = k0..+7 and k0+16..+23
            b.p[0] = *(const v8h*)(wm + N * LDH + k0);
            b.p[1] = *(const v8h*)(wm + N * LDH + k0 + 16);
            acc[t] = __builtin_amdgcn_wmma_f32_16x16x32_f16(
                false, a.v, false, b.v, (short)0, acc[t], false, false);
        }
    }
}

__device__ __forceinline__ void layer_norm(v8f (&h)[4], const float (&g)[4],
                                           const float (&be)[4]) {
#pragma unroll
    for (int r = 0; r < 8; ++r) {
        float s = h[0][r] + h[1][r] + h[2][r] + h[3][r];
        float q = h[0][r] * h[0][r] + h[1][r] * h[1][r] +
                  h[2][r] * h[2][r] + h[3][r] * h[3][r];
        s = bfly_add<SWZ_X1>(s);  q = bfly_add<SWZ_X1>(q);
        s = bfly_add<SWZ_X2>(s);  q = bfly_add<SWZ_X2>(q);
        s = bfly_add<SWZ_X4>(s);  q = bfly_add<SWZ_X4>(q);
        s = bfly_add<SWZ_X8>(s);  q = bfly_add<SWZ_X8>(q);
        float mu  = s * (1.0f / 64.0f);
        float var = q * (1.0f / 64.0f) - mu * mu;
        float inv = rsqrtf(var + 1e-5f);
#pragma unroll
        for (int t = 0; t < 4; ++t)
            h[t][r] = (h[t][r] - mu) * inv * g[t] + be[t];
    }
}

// One-time: convert W1/W2/W3 f32 -> f16 into d_ws, pre-padded to LDH stride,
// zone-major so each zone's weights are one contiguous 27648-B block for the TDM.
extern "C" __global__ void __launch_bounds__(256)
prep_weights_kernel(const float* __restrict__ W1, const float* __restrict__ W2,
                    const float* __restrict__ W3, _Float16* __restrict__ ws) {
    const int total = 3 * ZZ * ZZ * ZZ / 4;    // float4 count = 196608
    for (int i = blockIdx.x * blockDim.x + threadIdx.x; i < total;
         i += gridDim.x * blockDim.x) {
        int z   = i / 3072;                    // 3072 float4 per zone
        int rem = i - z * 3072;
        int w   = rem >> 10;
        int r2  = rem & 1023;
        int row = r2 >> 4;
        int c4  = (r2 & 15) << 2;
        const float* S = (w == 0 ? W1 : (w == 1 ? W2 : W3)) +
                         (size_t)z * ZZ * ZZ + row * ZZ + c4;
        const float4 v = *(const float4*)S;
        v4h p = {(_Float16)v.x, (_Float16)v.y, (_Float16)v.z, (_Float16)v.w};
        *(v4h*)(ws + ((size_t)z * 3 * ZZ + (size_t)w * ZZ + row) * LDH + c4) = p;
    }
}

extern "C" __global__ void __launch_bounds__(128)
zone_chain_kernel(const float* __restrict__ x, const float* __restrict__ Mask,
                  const float* __restrict__ b1, const float* __restrict__ g1,
                  const float* __restrict__ be1, const float* __restrict__ b2,
                  const float* __restrict__ g2, const float* __restrict__ be2,
                  const float* __restrict__ b3, const _Float16* __restrict__ ws,
                  float* __restrict__ out) {
    __shared__ __align__(16) _Float16 wbuf[2][WELEM];              // 55296 B, double-buffered
    __shared__ __align__(16) _Float16 hstage[NWAVES][WROWS * LDH]; // 9216 B

    const int tid  = threadIdx.x;
    const int wave = tid >> 5;
    const int lane = tid & 31;
    const int l    = lane & 15;
    const int half = lane >> 4;
    const long row_base = (long)blockIdx.x * BROWS + (long)wave * WROWS;

    _Float16* hst = hstage[wave];

    // Load x tile into C/D-layout registers
    v8f h[4];
#pragma unroll
    for (int t = 0; t < 4; ++t) {
#pragma unroll
        for (int r = 0; r < 8; ++r) {
            long m = row_base + r + 8 * half;
            h[t][r] = x[m * ZZ + t * 16 + l];
        }
    }

    // Prologue: bring zone 0 weights into buffer 0
#if defined(HAVE_TDM)
    if (wave == 0) {
        tdm_load_weights(ws, wbuf[0]);
        __builtin_amdgcn_s_wait_tensorcnt(0);
    }
#else
    coop_copy_weights(ws, wbuf[0], tid);
#endif
    __syncthreads();

    for (int z = 0; z < ZZ; ++z) {
        const _Float16* wb = wbuf[z & 1];

#if defined(HAVE_TDM)
        // Prefetch zone z+1 weights into the other buffer, overlapped with compute
        if (wave == 0 && z + 1 < ZZ)
            tdm_load_weights(ws + (size_t)(z + 1) * WELEM, wbuf[(z + 1) & 1]);
#endif

        // Per-lane zone parameters for this lane's 4 feature columns (L2-resident)
        float mk[4], BB1[4], BB2[4], BB3[4], G1v[4], BE1v[4], G2v[4], BE2v[4];
#pragma unroll
        for (int t = 0; t < 4; ++t) {
            int n  = z * ZZ + t * 16 + l;
            mk[t]  = Mask[n];
            BB1[t] = b1[n];  G1v[t] = g1[n];  BE1v[t] = be1[n];
            BB2[t] = b2[n];  G2v[t] = g2[n];  BE2v[t] = be2[n];
            BB3[t] = b3[n];
        }

        v8f acc[4];

        // layer 1: tanh((h*mask) @ W1^T + b1) -> LN
        stage_h<true>(hst, h, mk, l, half);
        wave_gemm(hst, wb + 0 * ZZ * LDH, l, half, BB1, acc);
#pragma unroll
        for (int t = 0; t < 4; ++t)
#pragma unroll
            for (int r = 0; r < 8; ++r) h[t][r] = fast_tanh(acc[t][r]);
        layer_norm(h, G1v, BE1v);

        // layer 2: tanh(h @ W2^T + b2) -> LN
        stage_h<false>(hst, h, mk, l, half);
        wave_gemm(hst, wb + 1 * ZZ * LDH, l, half, BB2, acc);
#pragma unroll
        for (int t = 0; t < 4; ++t)
#pragma unroll
            for (int r = 0; r < 8; ++r) h[t][r] = fast_tanh(acc[t][r]);
        layer_norm(h, G2v, BE2v);

        // layer 3: sigmoid(h @ W3^T + b3); becomes next zone's input
        stage_h<false>(hst, h, mk, l, half);
        wave_gemm(hst, wb + 2 * ZZ * LDH, l, half, BB3, acc);
#pragma unroll
        for (int t = 0; t < 4; ++t)
#pragma unroll
            for (int r = 0; r < 8; ++r) h[t][r] = fast_sigmoid(acc[t][r]);

        // store out[m][z][n] non-temporally (write-once, 256 MB >> L2)
#pragma unroll
        for (int t = 0; t < 4; ++t) {
#pragma unroll
            for (int r = 0; r < 8; ++r) {
                long m = row_base + r + 8 * half;
                __builtin_nontemporal_store(h[t][r], &out[(m * ZZ + z) * ZZ + t * 16 + l]);
            }
        }

#if defined(HAVE_TDM)
        if (wave == 0) __builtin_amdgcn_s_wait_tensorcnt(0);  // z+1 weights landed
        __syncthreads();                                      // visible to all waves
#else
        __syncthreads();
        if (z + 1 < ZZ) {
            coop_copy_weights(ws + (size_t)(z + 1) * WELEM, wbuf[(z + 1) & 1], tid);
            __syncthreads();
        }
#endif
    }
}

extern "C" void kernel_launch(void* const* d_in, const int* in_sizes, int n_in,
                              void* d_out, int out_size, void* d_ws, size_t ws_size,
                              hipStream_t stream) {
    const float* x    = (const float*)d_in[0];
    // d_in[1] = dtPt scalar (unused; derived from in_sizes)
    const float* Mask = (const float*)d_in[2];
    const float* W1   = (const float*)d_in[3];
    const float* b1   = (const float*)d_in[4];
    const float* g1   = (const float*)d_in[5];
    const float* be1  = (const float*)d_in[6];
    const float* W2   = (const float*)d_in[7];
    const float* b2   = (const float*)d_in[8];
    const float* g2   = (const float*)d_in[9];
    const float* be2  = (const float*)d_in[10];
    const float* W3   = (const float*)d_in[11];
    const float* b3   = (const float*)d_in[12];
    float* out = (float*)d_out;
    _Float16* ws = (_Float16*)d_ws;            // needs 3*64*64*72*2 = 1.77 MB

    // 1) convert+pad all zone weights to f16 in d_ws (one-time per call)
    hipLaunchKernelGGL(prep_weights_kernel, dim3(192), dim3(256), 0, stream,
                       W1, W2, W3, ws);

    // 2) zone chain: 256 blocks x 128 threads (4 waves, 16 rows each) = 1024 waves
    int nrows  = in_sizes[0] / ZZ;             // 16384
    int blocks = nrows / BROWS;                // 256
    hipLaunchKernelGGL(zone_chain_kernel, dim3(blocks), dim3(128), 0, stream,
                       x, Mask, b1, g1, be1, b2, g2, be2, b3, ws, out);
}
